// MultiHeadCasualAttention_3822520893933
// MI455X (gfx1250) — compile-verified, matrix operand order checked by reading, not run on hardware
//
#include <hip/hip_runtime.h>
#include <hip/hip_bf16.h>
#include <math.h>

// ---------------------------------------------------------------------------
// MI455X (gfx1250) fused multi-head causal attention block.
//   tokens (2,2048,1024) f32 -> out/k/v f32
// ~69 GFLOP dense, ~150-200 MB HBM traffic @ 23.3 TB/s -> compute & BW both
// O(10us).  bf16 WMMA (v_wmma_f32_16x16x32_bf16) for all matmuls, flash-style
// streaming softmax.  Data movement uses the CDNA5 async paths:
//   - global_load_async_to_lds_b128 (ASYNCcnt) for cooperative tile staging
//   - tensor_load_to_lds (TDM, TENSORcnt) for the GEMM B tiles
// Double-buffered LDS so the async engines overlap the WMMA pipeline.
// wave32, 8 waves / 256-thread block.
// ---------------------------------------------------------------------------

#define EMB   1024
#define NHEAD 16
#define DHEAD 64
#define LSEQ  2048
#define NBATCH 2
#define NROWS (NBATCH * LSEQ)   /* 4096 */

typedef __attribute__((ext_vector_type(16))) __bf16        v16bf;
typedef __attribute__((ext_vector_type(8)))  float         v8f;
typedef __attribute__((ext_vector_type(4)))  float         f32x4;
typedef __attribute__((ext_vector_type(4)))  unsigned int  u32x4;
typedef __attribute__((ext_vector_type(2)))  unsigned int  u32x2;
typedef __attribute__((ext_vector_type(4)))  int           i32x4;
typedef __attribute__((ext_vector_type(8)))  int           i32x8;

union FragU { u32x4 q[2]; v16bf v; };

__device__ __forceinline__ unsigned short f2bf(float f) {
  unsigned int u = __float_as_uint(f);
  unsigned int r = u + 0x7FFFu + ((u >> 16) & 1u);   // round-to-nearest-even
  return (unsigned short)(r >> 16);
}
__device__ __forceinline__ unsigned int pack2(float a, float b) {
  return (unsigned int)f2bf(a) | ((unsigned int)f2bf(b) << 16);
}

// 16x32 bf16 fragment load, row-major source, per CDNA5 16-bit A/B layout:
// lane<16  -> row = row16+lane,    K chunks [k..k+7] and [k+16..k+23]
// lane>=16 -> row = row16+lane-16, K chunks [k+8..k+15] and [k+24..k+31]
__device__ __forceinline__ v16bf load_frag(const unsigned short* base, int ld,
                                           int row16, int kbase, int lane) {
  int r = row16 + (lane & 15);
  int k = kbase + ((lane >> 4) << 3);
  const unsigned short* p = base + (size_t)r * ld + k;
  FragU u;
  u.q[0] = *(const u32x4*)(p);
  u.q[1] = *(const u32x4*)(p + 16);
  return u.v;
}

__device__ __forceinline__ v8f wmma_bf16(v16bf a, v16bf b, v8f c) {
  return __builtin_amdgcn_wmma_f32_16x16x32_bf16(
      /*neg_a=*/false, a, /*neg_b=*/false, b,
      /*c_mod=*/(short)0, c, /*reuse_a=*/false, /*reuse_b=*/false);
}

// Per-lane async copy of 16 bytes global -> LDS (tracked by ASYNCcnt).
// LDS dest address = low 32 bits of the generic pointer (LDS aperture maps
// addr[31:0] to the workgroup-relative LDS offset).
__device__ __forceinline__ void async_cp16(void* lds_ptr, const void* gptr) {
  unsigned l = (unsigned)(size_t)lds_ptr;
  asm volatile("global_load_async_to_lds_b128 %0, %1, off"
               :: "v"(l), "v"(gptr) : "memory");
}
__device__ __forceinline__ void wait_async0() {
  asm volatile("s_wait_asynccnt 0x0" ::: "memory");
}

// ---------------------------------------------------------------------------
// LayerNorm over a 1024-wide row; emits up to three (gamma,beta) variants as
// bf16.  One block per row, 256 threads x 4 elements.
// ---------------------------------------------------------------------------
__global__ __launch_bounds__(256) void ln_kernel(
    const float* __restrict__ x,
    const float* __restrict__ g0, const float* __restrict__ b0, unsigned short* __restrict__ o0,
    const float* __restrict__ g1, const float* __restrict__ b1, unsigned short* __restrict__ o1,
    const float* __restrict__ g2, const float* __restrict__ b2, unsigned short* __restrict__ o2)
{
  int row = blockIdx.x;
  int t = threadIdx.x;
  const float* xr = x + (size_t)row * 1024;
  f32x4 xl = *(const f32x4*)(xr + t * 4);
  float s  = xl.x + xl.y + xl.z + xl.w;
  float s2 = xl.x*xl.x + xl.y*xl.y + xl.z*xl.z + xl.w*xl.w;
#pragma unroll
  for (int o = 16; o >= 1; o >>= 1) {
    s  += __shfl_xor(s,  o, 32);
    s2 += __shfl_xor(s2, o, 32);
  }
  __shared__ float sh[16];
  int wid = t >> 5, lane = t & 31;
  if (lane == 0) { sh[wid] = s; sh[8 + wid] = s2; }
  __syncthreads();
  if (t == 0) {
    float a = 0.f, b = 0.f;
#pragma unroll
    for (int i = 0; i < 8; ++i) { a += sh[i]; b += sh[8 + i]; }
    sh[0] = a; sh[8] = b;
  }
  __syncthreads();
  float mean = sh[0] * (1.0f / 1024.0f);
  float var  = sh[8] * (1.0f / 1024.0f) - mean * mean;
  float rs   = rsqrtf(var + 1e-5f);
  float y0 = (xl.x - mean) * rs;
  float y1 = (xl.y - mean) * rs;
  float y2 = (xl.z - mean) * rs;
  float y3 = (xl.w - mean) * rs;
  int c = t * 4;
  size_t ob = (size_t)row * 1024 + c;
  {
    f32x4 g = *(const f32x4*)(g0 + c);
    f32x4 b = *(const f32x4*)(b0 + c);
    u32x2 w; w.x = pack2(y0*g.x + b.x, y1*g.y + b.y);
             w.y = pack2(y2*g.z + b.z, y3*g.w + b.w);
    *(u32x2*)(o0 + ob) = w;
  }
  if (o1) {
    f32x4 g = *(const f32x4*)(g1 + c);
    f32x4 b = *(const f32x4*)(b1 + c);
    u32x2 w; w.x = pack2(y0*g.x + b.x, y1*g.y + b.y);
             w.y = pack2(y2*g.z + b.z, y3*g.w + b.w);
    *(u32x2*)(o1 + ob) = w;
  }
  if (o2) {
    f32x4 g = *(const f32x4*)(g2 + c);
    f32x4 b = *(const f32x4*)(b2 + c);
    u32x2 w; w.x = pack2(y0*g.x + b.x, y1*g.y + b.y);
             w.y = pack2(y2*g.z + b.z, y3*g.w + b.w);
    *(u32x2*)(o2 + ob) = w;
  }
}

// ---------------------------------------------------------------------------
// Transpose+convert 1024x1024 f32 weight W(K,N) -> bf16 Wt(N,K) so WMMA
// B-fragments are contiguous along K (and TDM rows are contiguous).
// ---------------------------------------------------------------------------
__global__ __launch_bounds__(256) void wconv_kernel(
    const float* __restrict__ W, unsigned short* __restrict__ Wt)
{
  unsigned int idx = blockIdx.x * 256u + threadIdx.x;   // = n*1024 + k
  unsigned int n = idx >> 10;
  unsigned int k = idx & 1023u;
  Wt[idx] = f2bf(W[(size_t)k * 1024 + n]);
}

// ---------------------------------------------------------------------------
// WMMA GEMM: Y(4096x1024) = Xln(bf16) @ Wt^T + bias, epilogue by mode:
//  0: Q  -> bf16 (N,H,L,Dh)
//  1: K  -> f32  (N,H,Dh,L) into d_out  + bf16 (N,H,L,Dh)
//  2: V  -> f32  (N,H,L,Dh) into d_out  + bf16 (N,H,Dh,L)  (V^T for P@V)
//  3: final -> f32 row-major + residual tokens
// Block tile 64(M) x 128(N), K-step 32, double-buffered LDS:
//   A tile 64x32  staged with per-lane global_load_async_to_lds_b128
//   B tile 128x32 staged with one TDM tensor_load_to_lds (wave 0)
// ---------------------------------------------------------------------------
__global__ __launch_bounds__(256) void gemm_kernel(
    const unsigned short* __restrict__ Xln,   // (4096, 1024) bf16
    const unsigned short* __restrict__ Wt,    // (1024, 1024) bf16, N-major
    const float* __restrict__ bias,           // (1024)
    int mode,
    float* __restrict__ f32_out,
    unsigned short* __restrict__ bf16_out,
    const float* __restrict__ resid)
{
  __shared__ __align__(16) unsigned short lA[2][64 * 32];
  __shared__ __align__(16) unsigned short lB[2][128 * 32];

  int t = threadIdx.x;
  int lane = t & 31;
  int wave = t >> 5;
  int mblock = blockIdx.x * 64;
  int nblock = blockIdx.y * 128;
  int mloc = (wave >> 1) * 16;
  int nloc = (wave & 1) * 64;

  // A tile: 64 rows x 32 cols bf16 = 256 x 16B chunks, one per thread
  int arow = t >> 2;
  int achk = (t & 3) * 8;

  auto issueA = [&](int buf, int k) {
    async_cp16(&lA[buf][arow * 32 + achk],
               Xln + (size_t)(mblock + arow) * 1024 + k + achk);
  };
  auto issueB = [&](int buf, int k) {
    if (wave == 0) {
      // TDM 2-D descriptor: tensor = Wt (1024x1024, 2B elems), tile 32x128
      unsigned long long ga =
          (unsigned long long)(size_t)(Wt + (size_t)nblock * 1024 + k);
      unsigned lb = (unsigned)(size_t)&lB[buf][0];
      i32x4 g0;
      g0[0] = 1;                                   // count=1 (valid user D#)
      g0[1] = (int)lb;                             // lds_addr
      g0[2] = (int)(unsigned)(ga & 0xffffffffu);   // global_addr[31:0]
      g0[3] = (int)(((ga >> 32) & 0x01ffffffu) | 0x80000000u); // [56:32]|type=2
      i32x8 g1;
      g1[0] = 1 << 16;      // data_size=1 (2 bytes); wg_mask=0 (no cluster)
      g1[1] = 1024 << 16;   // tensor_dim0 = 1024   (bits 79:48)
      g1[2] = 1024 << 16;   // tensor_dim1 = 1024   (bits 111:80)
      g1[3] = 32 << 16;     // tile_dim0   = 32     (bits 127:112)
      g1[4] = 128;          // tile_dim1   = 128    (bits 143:128)
      g1[5] = 1024;         // tensor_dim0_stride = 1024 (bits 207:160)
      g1[6] = 0;
      g1[7] = 0;
      asm volatile("tensor_load_to_lds %0, %1" :: "s"(g0), "s"(g1) : "memory");
    }
  };

  v8f c0 = {}, c1 = {}, c2 = {}, c3 = {};
  issueA(0, 0);
  issueB(0, 0);
  for (int ks = 0; ks < 32; ++ks) {
    int buf = ks & 1;
    wait_async0();
    if (wave == 0) __builtin_amdgcn_s_wait_tensorcnt((short)0);
    __syncthreads();                       // publish tile ks, retire reads ks-1
    if (ks + 1 < 32) {                     // prefetch next tile into other buf
      issueA(buf ^ 1, (ks + 1) * 32);
      issueB(buf ^ 1, (ks + 1) * 32);
    }
    v16bf a = load_frag(lA[buf], 32, mloc, 0, lane);
    v16bf b;
    b = load_frag(lB[buf], 32, nloc +  0, 0, lane); c0 = wmma_bf16(a, b, c0);
    b = load_frag(lB[buf], 32, nloc + 16, 0, lane); c1 = wmma_bf16(a, b, c1);
    b = load_frag(lB[buf], 32, nloc + 32, 0, lane); c2 = wmma_bf16(a, b, c2);
    b = load_frag(lB[buf], 32, nloc + 48, 0, lane); c3 = wmma_bf16(a, b, c3);
  }

  int half = lane >> 4;
  int ncol = lane & 15;
  int mbase = mblock + mloc;
  int nbase = nblock + nloc;
  v8f cc[4] = { c0, c1, c2, c3 };
#pragma unroll
  for (int j = 0; j < 4; ++j) {
    int n = nbase + j * 16 + ncol;
    float bn = bias[n];
    int h = n >> 6;          // head
    int d = n & 63;          // dh
#pragma unroll
    for (int r = 0; r < 8; ++r) {
      int m = mbase + r + half * 8;          // global row in [0,4096)
      float y = cc[j][r] + bn;
      int nb = m >> 11;                      // batch
      int l  = m & (LSEQ - 1);
      size_t nh = (size_t)(nb * NHEAD + h);
      if (mode == 0) {
        bf16_out[(nh * LSEQ + l) * DHEAD + d] = f2bf(y);
      } else if (mode == 1) {
        f32_out[(nh * DHEAD + d) * LSEQ + l] = y;
        bf16_out[(nh * LSEQ + l) * DHEAD + d] = f2bf(y);
      } else if (mode == 2) {
        f32_out[(nh * LSEQ + l) * DHEAD + d] = y;
        bf16_out[(nh * DHEAD + d) * LSEQ + l] = f2bf(y);
      } else {
        size_t o = (size_t)m * 1024 + n;
        f32_out[o] = y + resid[o];
      }
    }
  }
}

// ---------------------------------------------------------------------------
// Flash attention: one (batch*head) per blockIdx.y, 128 query rows per block,
// 16 rows per wave.  All waves iterate a uniform key range (early waves run
// fully-masked iterations, numerically inert) so K/V tiles are staged
// cooperatively into double-buffered LDS with async loads (8x L2 traffic cut).
// P is staged through a wave-private LDS tile to convert C-layout->A-layout.
// ---------------------------------------------------------------------------
__global__ __launch_bounds__(256) void attn_kernel(
    const unsigned short* __restrict__ Qb,   // (N*H, L, Dh) bf16
    const unsigned short* __restrict__ Kb,   // (N*H, L, Dh) bf16
    const unsigned short* __restrict__ Vt,   // (N*H, Dh, L) bf16
    float* __restrict__ O)                   // (N, L, H*Dh) f32
{
  __shared__ __align__(16) unsigned short lK[2][32 * 64];   // keys x dh
  __shared__ __align__(16) unsigned short lV[2][64 * 32];   // dh x keys
  __shared__ __align__(16) unsigned short pstage[8][16 * 32];

  int t = threadIdx.x;
  int lane = t & 31;
  int wave = t >> 5;
  int nh   = blockIdx.y;                  // n*H + h
  int qb0  = blockIdx.x * 128;
  int qb   = qb0 + wave * 16;

  const unsigned short* q  = Qb + (size_t)nh * LSEQ * DHEAD;
  const unsigned short* kk = Kb + (size_t)nh * LSEQ * DHEAD;
  const unsigned short* vt = Vt + (size_t)nh * DHEAD * LSEQ;

  v16bf aq0 = load_frag(q, DHEAD, qb,  0, lane);
  v16bf aq1 = load_frag(q, DHEAD, qb, 32, lane);

  // staging maps: K tile 32x64 (8 chunks/row), V tile 64x32 (4 chunks/row)
  int krow = t >> 3, kchk = (t & 7) * 8;
  int vrow = t >> 2, vchk = (t & 3) * 8;
  auto issue = [&](int buf, int j0) {
    async_cp16(&lK[buf][krow * 64 + kchk],
               kk + (size_t)(j0 + krow) * DHEAD + kchk);
    async_cp16(&lV[buf][vrow * 32 + vchk],
               vt + (size_t)vrow * LSEQ + j0 + vchk);
  };

  v8f o0 = {}, o1 = {}, o2 = {}, o3 = {};
  float mrow[8], srow[8];
#pragma unroll
  for (int r = 0; r < 8; ++r) { mrow[r] = -3.0e38f; srow[r] = 0.f; }

  int half = lane >> 4;
  int ncol = lane & 15;
  unsigned short* pl = pstage[wave];
  const float sc = 0.125f;                // 1/sqrt(64)

  int nIter = blockIdx.x * 4 + 4;         // covers keys 0 .. qb0+127
  issue(0, 0);
  for (int it = 0; it < nIter; ++it) {
    int j0 = it * 32;
    int buf = it & 1;
    wait_async0();
    __syncthreads();                      // publish tile it, retire reads it-1
    if (it + 1 < nIter) issue(buf ^ 1, j0 + 32);

    // ---- S = Q K^T for 32 keys: two 16x16 tiles, K=64 each
    v8f s0 = {}, s1 = {};
    {
      v16bf bk;
      bk = load_frag(lK[buf], DHEAD,  0,  0, lane); s0 = wmma_bf16(aq0, bk, s0);
      bk = load_frag(lK[buf], DHEAD,  0, 32, lane); s0 = wmma_bf16(aq1, bk, s0);
      bk = load_frag(lK[buf], DHEAD, 16,  0, lane); s1 = wmma_bf16(aq0, bk, s1);
      bk = load_frag(lK[buf], DHEAD, 16, 32, lane); s1 = wmma_bf16(aq1, bk, s1);
    }
    // ---- mask (+-9999 before /sqrt(Dh), as reference) + online softmax
    float rescale[8];
#pragma unroll
    for (int r = 0; r < 8; ++r) {
      int i  = qb + r + half * 8;
      int ja = j0 + ncol;
      float va = (s0[r] + ((ja      > i) ? -9999.0f : 0.0f)) * sc;
      float vb = (s1[r] + ((ja + 16 > i) ? -9999.0f : 0.0f)) * sc;
      float mx = fmaxf(va, vb);
      mx = fmaxf(mx, __shfl_xor(mx, 1, 32));
      mx = fmaxf(mx, __shfl_xor(mx, 2, 32));
      mx = fmaxf(mx, __shfl_xor(mx, 4, 32));
      mx = fmaxf(mx, __shfl_xor(mx, 8, 32));
      float mnew = fmaxf(mrow[r], mx);
      float al = __expf(mrow[r] - mnew);
      float p0 = __expf(va - mnew);
      float p1 = __expf(vb - mnew);
      s0[r] = p0; s1[r] = p1;
      float ps = p0 + p1;
      ps += __shfl_xor(ps, 1, 32);
      ps += __shfl_xor(ps, 2, 32);
      ps += __shfl_xor(ps, 4, 32);
      ps += __shfl_xor(ps, 8, 32);
      srow[r] = srow[r] * al + ps;
      mrow[r] = mnew;
      rescale[r] = al;
    }
#pragma unroll
    for (int r = 0; r < 8; ++r) {
      o0[r] *= rescale[r]; o1[r] *= rescale[r];
      o2[r] *= rescale[r]; o3[r] *= rescale[r];
    }
    // ---- stage P (16x32) as bf16 in wave-private LDS (C-layout -> row major)
#pragma unroll
    for (int r = 0; r < 8; ++r) {
      int m = r + half * 8;
      pl[m * 32 + ncol]      = f2bf(s0[r]);
      pl[m * 32 + 16 + ncol] = f2bf(s1[r]);
    }
    // cross-lane LDS RAW within the wave: DS ops are in-order per wave, but
    // force HW wait + stop the compiler reordering the b128 reloads.
    asm volatile("s_wait_dscnt 0x0" ::: "memory");
    v16bf pa = load_frag(pl, 32, 0, 0, lane);
    // ---- O += P @ V  (B = V^T rows = dh, contiguous over keys)
    {
      v16bf bv;
      bv = load_frag(lV[buf], 32,  0, 0, lane); o0 = wmma_bf16(pa, bv, o0);
      bv = load_frag(lV[buf], 32, 16, 0, lane); o1 = wmma_bf16(pa, bv, o1);
      bv = load_frag(lV[buf], 32, 32, 0, lane); o2 = wmma_bf16(pa, bv, o2);
      bv = load_frag(lV[buf], 32, 48, 0, lane); o3 = wmma_bf16(pa, bv, o3);
    }
  }

  // ---- normalize and scatter to (N, L, H*Dh)
  int n = nh >> 4;
  int h = nh & 15;
  v8f oo[4] = { o0, o1, o2, o3 };
#pragma unroll
  for (int r = 0; r < 8; ++r) {
    float inv = 1.0f / srow[r];
    int m = qb + r + half * 8;
    size_t rowoff = ((size_t)n * LSEQ + m) * 1024 + h * DHEAD;
#pragma unroll
    for (int f = 0; f < 4; ++f)
      O[rowoff + f * 16 + ncol] = oo[f][r] * inv;
  }
}

// ---------------------------------------------------------------------------
extern "C" void kernel_launch(void* const* d_in, const int* in_sizes, int n_in,
                              void* d_out, int out_size, void* d_ws, size_t ws_size,
                              hipStream_t stream) {
  const float* tokens = (const float*)d_in[0];
  const float* ln_q_g = (const float*)d_in[1];
  const float* ln_q_b = (const float*)d_in[2];
  const float* Wq     = (const float*)d_in[3];
  const float* bq     = (const float*)d_in[4];
  const float* ln_k_g = (const float*)d_in[5];
  const float* ln_k_b = (const float*)d_in[6];
  const float* Wk     = (const float*)d_in[7];
  const float* bk     = (const float*)d_in[8];
  const float* ln_v_g = (const float*)d_in[9];
  const float* ln_v_b = (const float*)d_in[10];
  const float* Wv     = (const float*)d_in[11];
  const float* bv     = (const float*)d_in[12];
  const float* ln_o_g = (const float*)d_in[13];
  const float* ln_o_b = (const float*)d_in[14];
  const float* Wo     = (const float*)d_in[15];
  const float* bo     = (const float*)d_in[16];

  char* wsp = (char*)d_ws;
  size_t off = 0;
  auto wsa = [&](size_t bytes) -> void* {
    void* p = wsp + off;
    off += (bytes + 255) & ~(size_t)255;
    return p;
  };
  const size_t actB = (size_t)NROWS * 1024 * sizeof(unsigned short); // 8 MiB
  const size_t wB   = (size_t)1024 * 1024 * sizeof(unsigned short);  // 2 MiB
  unsigned short* lnQ = (unsigned short*)wsa(actB);  // LN w/ ln_q
  unsigned short* lnK = (unsigned short*)wsa(actB);  // LN w/ ln_v  (k path)
  unsigned short* lnV = (unsigned short*)wsa(actB);  // LN w/ ln_k  (v path)
  unsigned short* wtQ = (unsigned short*)wsa(wB);    // Wq^T
  unsigned short* wtK = (unsigned short*)wsa(wB);    // Wv^T
  unsigned short* wtV = (unsigned short*)wsa(wB);    // Wk^T
  unsigned short* wtO = (unsigned short*)wsa(wB);    // Wo^T
  unsigned short* qbf = (unsigned short*)wsa(actB);  // (N,H,L,Dh)
  unsigned short* kbf = (unsigned short*)wsa(actB);  // (N,H,L,Dh)
  unsigned short* vtb = (unsigned short*)wsa(actB);  // (N,H,Dh,L)
  float*          oac = (float*)wsa((size_t)NROWS * 1024 * sizeof(float));
  unsigned short* lnO = (unsigned short*)wsa(actB);
  (void)ws_size; (void)in_sizes; (void)n_in; (void)out_size;

  float* out_o = (float*)d_out;
  float* out_k = out_o + (size_t)NROWS * 1024;   // (N,H,Dh,L)
  float* out_v = out_k + (size_t)NROWS * 1024;   // (N,H,L,Dh)

  // 1) three LayerNorms of tokens (reference quirk: K uses ln_v, V uses ln_k)
  ln_kernel<<<NROWS, 256, 0, stream>>>(tokens,
      ln_q_g, ln_q_b, lnQ,
      ln_v_g, ln_v_b, lnK,
      ln_k_g, ln_k_b, lnV);

  // 2) weight transpose -> bf16 (N-major)
  wconv_kernel<<<4096, 256, 0, stream>>>(Wq, wtQ);
  wconv_kernel<<<4096, 256, 0, stream>>>(Wv, wtK);
  wconv_kernel<<<4096, 256, 0, stream>>>(Wk, wtV);
  wconv_kernel<<<4096, 256, 0, stream>>>(Wo, wtO);

  // 3) projections (WMMA GEMMs, async+TDM staged) with layout epilogues
  dim3 ggrid(NROWS / 64, 1024 / 128);
  gemm_kernel<<<ggrid, 256, 0, stream>>>(lnQ, wtQ, bq, 0, nullptr, qbf, nullptr);
  gemm_kernel<<<ggrid, 256, 0, stream>>>(lnK, wtK, bv, 1, out_k, kbf, nullptr);
  gemm_kernel<<<ggrid, 256, 0, stream>>>(lnV, wtV, bk, 2, out_v, vtb, nullptr);

  // 4) causal flash attention (async-staged K/V)
  attn_kernel<<<dim3(LSEQ / 128, NBATCH * NHEAD), 256, 0, stream>>>(qbf, kbf, vtb, oac);

  // 5) output LayerNorm + projection + residual
  ln_kernel<<<NROWS, 256, 0, stream>>>(oac,
      ln_o_g, ln_o_b, lnO,
      nullptr, nullptr, nullptr,
      nullptr, nullptr, nullptr);
  gemm_kernel<<<ggrid, 256, 0, stream>>>(lnO, wtO, bo, 3, out_o, nullptr, tokens);
}
